// MultiHeadSelfAttention_28054726378086
// MI455X (gfx1250) — matrix operand order repaired and checked by reading, not compiled, not run on hardware
//
#include <hip/hip_runtime.h>
#include <cstdint>
#include <cstddef>

// ---------------------------------------------------------------------------
// MI455X (gfx1250, wave32) fused multi-head attention
//   B=4, N=2048, E=1024, H=16, D=64
// All GEMMs via v_wmma_f32_16x16x32_f16 (f32 accumulate). Inputs/weights are
// pre-converted to f16 once so GEMM hot loops are pure b128-load -> wmma.
// Attention K/V tiles staged into LDS by the Tensor Data Mover
// (tensor_load_to_lds), double-buffered with s_wait_tensorcnt.
// ---------------------------------------------------------------------------

typedef __attribute__((ext_vector_type(16))) _Float16     v16h;
typedef __attribute__((ext_vector_type(8)))  float        v8f;
typedef __attribute__((ext_vector_type(4)))  unsigned int v4u;
typedef __attribute__((ext_vector_type(8)))  int          v8i;
typedef __attribute__((ext_vector_type(4)))  int          v4i;

union Frag16 {
    v16h     v;
    _Float16 h[16];
    uint4    q[2];   // two 16-byte halves: elements 0..7 / 8..15
};

__device__ __forceinline__ v8f v8f_zero() {
    v8f z;
#pragma unroll
    for (int i = 0; i < 8; ++i) z[i] = 0.0f;
    return z;
}

__device__ __forceinline__ v8f wmma16(const Frag16& a, const Frag16& b, v8f c) {
    // 8 args: (neg_a, A, neg_b, B, c_mod, C, reuse_a, reuse_b)
    return __builtin_amdgcn_wmma_f32_16x16x32_f16(false, a.v, false, b.v,
                                                  (short)0, c, false, false);
}

#define E_DIM 1024
#define N_DIM 2048
#define H_DIM 16
#define D_DIM 64

// ---------------------------------------------------------------------------
// Kernel 0: bulk f32 -> f16 conversion (vectorized, 8 elems/thread).
// ---------------------------------------------------------------------------
__global__ __launch_bounds__(256) void f32_to_f16_kernel(
    const float* __restrict__ src, _Float16* __restrict__ dst, int n8)
{
    const int i = blockIdx.x * 256 + threadIdx.x;
    if (i >= n8) return;
    const float4* s = (const float4*)(src + (size_t)i * 8);
    const float4 f0 = s[0];
    const float4 f1 = s[1];
    union { _Float16 h[8]; uint4 u; } o;
    o.h[0] = (_Float16)f0.x; o.h[1] = (_Float16)f0.y;
    o.h[2] = (_Float16)f0.z; o.h[3] = (_Float16)f0.w;
    o.h[4] = (_Float16)f1.x; o.h[5] = (_Float16)f1.y;
    o.h[6] = (_Float16)f1.z; o.h[7] = (_Float16)f1.w;
    *(uint4*)(dst + (size_t)i * 8) = o.u;
}

// ---------------------------------------------------------------------------
// TDM: async 2-D tile load (f16 elements) global -> LDS.
// Descriptor packed per CDNA5 ISA ch.8 (D# group0/group1).
// tensor dims == tile dims (tiles never OOB: 2048 % 32 == 0), strides are the
// true row strides of the source tensor. 6-arg builtin on this toolchain:
//   (uint32x4 g0, int32x8 g1, int32x4 g2, int32x4 g3, int32x8 pad, i32 cpol)
// ---------------------------------------------------------------------------
__device__ __forceinline__ void tdm_load_2d_f16(
    unsigned lds_off, const _Float16* gptr,
    unsigned tile_w, unsigned tile_h, unsigned long long row_stride_elems)
{
    unsigned long long ga = (unsigned long long)(uintptr_t)gptr;  // bytes
    v4u g0;
    g0[0] = 1u;                                        // count=1 (valid), load
    g0[1] = lds_off;                                   // LDS byte address
    g0[2] = (unsigned)(ga & 0xFFFFFFFFu);              // global_addr[31:0]
    g0[3] = (unsigned)((ga >> 32) & 0x1FFFFFFu)        // global_addr[56:32]
          | (2u << 30);                                // type = 2 ("image")

    v8i g1;
    g1[0] = (int)(1u << 16);                           // data_size=1 -> 2 bytes
    g1[1] = (int)((tile_w & 0xFFFFu) << 16);           // tensor_dim0[15:0]
    g1[2] = (int)((tile_h & 0xFFFFu) << 16);           // tensor_dim1[15:0]
    g1[3] = (int)((tile_w & 0xFFFFu) << 16);           // tile_dim0
    g1[4] = (int)(tile_h & 0xFFFFu);                   // tile_dim1 (tile_dim2=0)
    g1[5] = (int)(unsigned)(row_stride_elems & 0xFFFFFFFFull);      // stride0 lo
    g1[6] = (int)(unsigned)((row_stride_elems >> 32) & 0xFFFFull);  // stride0 hi
    g1[7] = 0;                                         // tensor_dim1_stride=0

    v4i z4; z4[0] = 0; z4[1] = 0; z4[2] = 0; z4[3] = 0;   // groups 2/3: <=2D
    v8i z8;
#pragma unroll
    for (int i = 0; i < 8; ++i) z8[i] = 0;
    __builtin_amdgcn_tensor_load_to_lds(g0, g1, z4, z4, z8, 0);
}

// ---------------------------------------------------------------------------
// Kernel 1: projection GEMM  Y[m,f] = sum_e X[m,e]*W[f,e] + b[f] (f16 in/out)
// One wave computes a 32x64 output block: 2 A-frags x 4 B-frags = 8 WMMAs
// per K-step; every fragment load is a contiguous global_load_b128.
// LAYOUT 0: out[b][h][npos][d]   (row-major per head; Q, K)
// LAYOUT 1: out[b][h][d][npos]   (transposed per head; V)
// ---------------------------------------------------------------------------
template <int LAYOUT>
__global__ __launch_bounds__(256) void qkv_proj_kernel(
    const _Float16* __restrict__ X, const _Float16* __restrict__ W,
    const float* __restrict__ bias, _Float16* __restrict__ out)
{
    const int lane = threadIdx.x & 31;
    const int wave = threadIdx.x >> 5;
    const int hlf  = lane >> 4;      // 0 or 1 (wave32 halves)
    const int l16  = lane & 15;

    const int tile_f = blockIdx.x;              // 0..15  (64-feature strips)
    const int tile_m = blockIdx.y * 8 + wave;   // 0..255 (32-row tiles)

    const _Float16* arow0 = X + (size_t)(tile_m * 32 + l16) * E_DIM;
    const _Float16* arow1 = arow0 + (size_t)16 * E_DIM;
    const _Float16* wrow[4];
#pragma unroll
    for (int sub = 0; sub < 4; ++sub)
        wrow[sub] = W + (size_t)(tile_f * 64 + sub * 16 + l16) * E_DIM;

    v8f acc[2][4];
#pragma unroll
    for (int g = 0; g < 2; ++g)
#pragma unroll
        for (int sub = 0; sub < 4; ++sub) acc[g][sub] = v8f_zero();

    for (int k0 = 0; k0 < E_DIM; k0 += 32) {
        if (k0 + 32 < E_DIM) {
            __builtin_prefetch(arow0 + k0 + 32, 0, 1);
            __builtin_prefetch(arow1 + k0 + 32, 0, 1);
        }
        // A fragments: row = lane&15; element j -> k = 16*(j/8)+8*hlf+(j&7)
        Frag16 a0, a1;
        a0.q[0] = *(const uint4*)(arow0 + k0 + hlf * 8);
        a0.q[1] = *(const uint4*)(arow0 + k0 + 16 + hlf * 8);
        a1.q[0] = *(const uint4*)(arow1 + k0 + hlf * 8);
        a1.q[1] = *(const uint4*)(arow1 + k0 + 16 + hlf * 8);

#pragma unroll
        for (int sub = 0; sub < 4; ++sub) {
            Frag16 b;   // col = lane&15; element j -> k = 16*(j/8)+8*hlf+(j&7)
            const _Float16* bs = wrow[sub] + k0 + hlf * 8;
            b.q[0] = *(const uint4*)(bs);
            b.q[1] = *(const uint4*)(bs + 16);
            acc[0][sub] = wmma16(a0, b, acc[0][sub]);
            acc[1][sub] = wmma16(a1, b, acc[1][sub]);
        }
    }

#pragma unroll
    for (int sub = 0; sub < 4; ++sub) {
        const int fcol = tile_f * 64 + sub * 16 + l16;
        const float bv = bias[fcol];
        const int hh = fcol >> 6;      // head
        const int dd = fcol & 63;      // dim within head
#pragma unroll
        for (int g = 0; g < 2; ++g) {
#pragma unroll
            for (int r = 0; r < 8; ++r) {
                const int mg   = tile_m * 32 + g * 16 + r + 8 * hlf;
                const int bb   = mg >> 11;                   // batch (N=2048)
                const int npos = mg & 2047;
                const float val = acc[g][sub][r] + bv;
                size_t addr;
                if (LAYOUT == 0)
                    addr = (((size_t)bb * H_DIM + hh) * N_DIM + npos) * D_DIM + dd;
                else
                    addr = (((size_t)bb * H_DIM + hh) * D_DIM + dd) * N_DIM + npos;
                out[addr] = (_Float16)val;
            }
        }
    }
}

// ---------------------------------------------------------------------------
// Kernel 2: flash attention. One block = 4 waves = 4 q-tiles of one (b,h).
// K/V 32-key blocks are TDM-loaded into double-buffered LDS tiles shared by
// all 4 waves; softmax is online (streaming max/sum).
// Q,K: [B,H,N,D] f16.  Vt: [B,H,D,N] f16.  AO: [B,N,E] f16.
// ---------------------------------------------------------------------------
__global__ __launch_bounds__(128) void attn_kernel(
    const _Float16* __restrict__ Q, const _Float16* __restrict__ K,
    const _Float16* __restrict__ Vt, _Float16* __restrict__ AO)
{
    const int lane = threadIdx.x & 31;
    const int wave = threadIdx.x >> 5;   // 0..3
    const int hlf  = lane >> 4;
    const int l16  = lane & 15;

    const int qt = blockIdx.x * 4 + wave;  // 0..127 (16-row q tile)
    const int hh = blockIdx.y;             // 0..15
    const int bb = blockIdx.z;             // 0..3

    const size_t head = (size_t)bb * H_DIM + hh;
    const _Float16* qh = Q  + head * N_DIM * D_DIM;
    const _Float16* kh = K  + head * N_DIM * D_DIM;
    const _Float16* vh = Vt + head * D_DIM * N_DIM;

    __shared__ _Float16 ldsK[2][32][64];   // [buf][key pos][d]       4KB each
    __shared__ _Float16 ldsV[2][64][32];   // [buf][d][key pos local] 4KB each
    __shared__ _Float16 pst[4][16][32];    // per-wave P staging

    const unsigned offK[2] = { (unsigned)(uintptr_t)&ldsK[0][0][0],
                               (unsigned)(uintptr_t)&ldsK[1][0][0] };
    const unsigned offV[2] = { (unsigned)(uintptr_t)&ldsV[0][0][0],
                               (unsigned)(uintptr_t)&ldsV[1][0][0] };

    // Q fragments: two K=32 chunks covering D=64
    Frag16 aq[2];
    const _Float16* qrow = qh + (size_t)(qt * 16 + l16) * D_DIM;
#pragma unroll
    for (int c = 0; c < 2; ++c) {
        aq[c].q[0] = *(const uint4*)(qrow + c * 32 + hlf * 8);
        aq[c].q[1] = *(const uint4*)(qrow + c * 32 + 16 + hlf * 8);
    }

    float mprev[8], lsum[8];
    v8f oacc[4];
#pragma unroll
    for (int r = 0; r < 8; ++r) { mprev[r] = -3.0e38f; lsum[r] = 0.0f; }
#pragma unroll
    for (int dt = 0; dt < 4; ++dt) oacc[dt] = v8f_zero();

    const float scale = 0.125f;   // D^-0.5 = 1/8

    // Prologue: TDM-load block 0 into buffer 0 (wave 0 drives the DMA)
    if (wave == 0) {
        tdm_load_2d_f16(offK[0], kh, /*w=*/64, /*h=*/32, /*stride=*/D_DIM);
        tdm_load_2d_f16(offV[0], vh, /*w=*/32, /*h=*/64, /*stride=*/N_DIM);
    }

    for (int kb = 0; kb < N_DIM; kb += 32) {
        const int buf = (kb >> 5) & 1;
        const bool hasNext = (kb + 32) < N_DIM;

        __syncthreads();   // everyone done reading buf^1 before it is rewritten
        if (wave == 0) {
            if (hasNext) {
                tdm_load_2d_f16(offK[buf ^ 1], kh + (size_t)(kb + 32) * D_DIM,
                                64, 32, D_DIM);
                tdm_load_2d_f16(offV[buf ^ 1], vh + (kb + 32),
                                32, 64, N_DIM);
                __builtin_amdgcn_s_wait_tensorcnt(2);  // current block landed
            } else {
                __builtin_amdgcn_s_wait_tensorcnt(0);
            }
        }
        __syncthreads();   // current buffer visible to all waves
        __asm__ volatile("" ::: "memory");

        // ---- scores: two 16-wide key sub-tiles, K-dim = D = 64 (2x32) ----
        v8f s[2];
#pragma unroll
        for (int t = 0; t < 2; ++t) {
            v8f sc = v8f_zero();
            const _Float16* krow = &ldsK[buf][t * 16 + l16][0];
#pragma unroll
            for (int c = 0; c < 2; ++c) {
                Frag16 bk;  // B frag: col = key pos (lane&15), k = d (table order)
                bk.q[0] = *(const uint4*)(krow + c * 32 + hlf * 8);
                bk.q[1] = *(const uint4*)(krow + c * 32 + 16 + hlf * 8);
                sc = wmma16(aq[c], bk, sc);
            }
            s[t] = sc;
        }

        // ---- scale + mask + online softmax (rows striped: m = r + 8*hlf) ----
        const int k0pos = kb + l16;
        const int k1pos = kb + 16 + l16;
#pragma unroll
        for (int r = 0; r < 8; ++r) {
            const int qpos = qt * 16 + r + 8 * hlf;
            float s0 = s[0][r] * scale;
            float s1 = s[1][r] * scale;
            if (qpos == 0 && k0pos == N_DIM - 1) s0 = -1e9f;
            if (qpos == 0 && k1pos == N_DIM - 1) s1 = -1e9f;
            if (qpos == N_DIM - 1 && k0pos == 0) s0 = -1e9f;
            if (qpos == N_DIM - 1 && k1pos == 0) s1 = -1e9f;

            float mx = fmaxf(s0, s1);
#pragma unroll
            for (int off = 1; off < 16; off <<= 1)
                mx = fmaxf(mx, __shfl_xor(mx, off, 16));
            const float mnew  = fmaxf(mprev[r], mx);
            const float alpha = __expf(mprev[r] - mnew);
            const float p0 = __expf(s0 - mnew);
            const float p1 = __expf(s1 - mnew);
            float rs = p0 + p1;
#pragma unroll
            for (int off = 1; off < 16; off <<= 1)
                rs += __shfl_xor(rs, off, 16);
            lsum[r]  = lsum[r] * alpha + rs;
            mprev[r] = mnew;
#pragma unroll
            for (int dt = 0; dt < 4; ++dt) oacc[dt][r] *= alpha;

            // stage P into LDS (row-major 16x32) for C-layout -> A-layout
            pst[wave][r + 8 * hlf][l16]      = (_Float16)p0;
            pst[wave][r + 8 * hlf][16 + l16] = (_Float16)p1;
        }
        __asm__ volatile("" ::: "memory");  // keep ds_load after ds_store

        // ---- read P back in A-fragment layout (16x32, K = key pos) ----
        Frag16 ap;
        ap.q[0] = *(const uint4*)(&pst[wave][l16][hlf * 8]);
        ap.q[1] = *(const uint4*)(&pst[wave][l16][16 + hlf * 8]);

        // ---- O += P * V : 4 d-tiles of 16 ----
#pragma unroll
        for (int dt = 0; dt < 4; ++dt) {
            const _Float16* vrow = &ldsV[buf][dt * 16 + l16][hlf * 8];
            Frag16 bv;  // B frag: col = d (lane&15), k = key pos (table order)
            bv.q[0] = *(const uint4*)(vrow);
            bv.q[1] = *(const uint4*)(vrow + 16);
            oacc[dt] = wmma16(ap, bv, oacc[dt]);
        }
    }

    // ---- normalize and store attention output as [B,N,E] f16 ----
#pragma unroll
    for (int r = 0; r < 8; ++r) {
        const float inv = 1.0f / lsum[r];
        const int m = qt * 16 + r + 8 * hlf;
        const size_t addr = ((size_t)bb * N_DIM + m) * E_DIM + hh * 64;
#pragma unroll
        for (int dt = 0; dt < 4; ++dt)
            AO[addr + dt * 16 + l16] = (_Float16)(oacc[dt][r] * inv);
    }
}

// ---------------------------------------------------------------------------
// Kernel 3: output projection  Y[m,f] = sum_e A[m,e]*Wo[f,e] (f16 in, f32 out)
// One wave computes a 32x64 block (8 WMMAs per K-step).
// ---------------------------------------------------------------------------
__global__ __launch_bounds__(256) void out_proj_kernel(
    const _Float16* __restrict__ A, const _Float16* __restrict__ W,
    float* __restrict__ out)
{
    const int lane = threadIdx.x & 31;
    const int wave = threadIdx.x >> 5;
    const int hlf  = lane >> 4;
    const int l16  = lane & 15;

    const int tile_f = blockIdx.x;              // 0..15
    const int tile_m = blockIdx.y * 8 + wave;   // 0..255

    const _Float16* arow0 = A + (size_t)(tile_m * 32 + l16) * E_DIM;
    const _Float16* arow1 = arow0 + (size_t)16 * E_DIM;
    const _Float16* wrow[4];
#pragma unroll
    for (int sub = 0; sub < 4; ++sub)
        wrow[sub] = W + (size_t)(tile_f * 64 + sub * 16 + l16) * E_DIM;

    v8f acc[2][4];
#pragma unroll
    for (int g = 0; g < 2; ++g)
#pragma unroll
        for (int sub = 0; sub < 4; ++sub) acc[g][sub] = v8f_zero();

    for (int k0 = 0; k0 < E_DIM; k0 += 32) {
        if (k0 + 32 < E_DIM) {
            __builtin_prefetch(arow0 + k0 + 32, 0, 1);
            __builtin_prefetch(arow1 + k0 + 32, 0, 1);
        }
        Frag16 a0, a1;
        a0.q[0] = *(const uint4*)(arow0 + k0 + hlf * 8);
        a0.q[1] = *(const uint4*)(arow0 + k0 + 16 + hlf * 8);
        a1.q[0] = *(const uint4*)(arow1 + k0 + hlf * 8);
        a1.q[1] = *(const uint4*)(arow1 + k0 + 16 + hlf * 8);

#pragma unroll
        for (int sub = 0; sub < 4; ++sub) {
            Frag16 b;
            const _Float16* bs = wrow[sub] + k0 + hlf * 8;
            b.q[0] = *(const uint4*)(bs);
            b.q[1] = *(const uint4*)(bs + 16);
            acc[0][sub] = wmma16(a0, b, acc[0][sub]);
            acc[1][sub] = wmma16(a1, b, acc[1][sub]);
        }
    }

#pragma unroll
    for (int sub = 0; sub < 4; ++sub) {
#pragma unroll
        for (int g = 0; g < 2; ++g) {
#pragma unroll
            for (int r = 0; r < 8; ++r) {
                const size_t m = (size_t)(tile_m * 32 + g * 16 + r + 8 * hlf);
                out[m * E_DIM + tile_f * 64 + sub * 16 + l16] = acc[g][sub][r];
            }
        }
    }
}

// ---------------------------------------------------------------------------
// Launch
// ---------------------------------------------------------------------------
extern "C" void kernel_launch(void* const* d_in, const int* in_sizes, int n_in,
                              void* d_out, int out_size, void* d_ws, size_t ws_size,
                              hipStream_t stream) {
    const float* x  = (const float*)d_in[0];
    const float* Wq = (const float*)d_in[1];
    const float* bq = (const float*)d_in[2];
    const float* Wk = (const float*)d_in[3];
    const float* bk = (const float*)d_in[4];
    const float* Wv = (const float*)d_in[5];
    const float* bv = (const float*)d_in[6];
    const float* Wo = (const float*)d_in[7];
    float* out = (float*)d_out;

    const size_t S = (size_t)4 * 2048 * 1024;   // activation f16 buffer elems
    const size_t WS = (size_t)1024 * 1024;      // weight f16 buffer elems
    _Float16* qbuf  = (_Float16*)d_ws;          // [B,H,N,D]
    _Float16* kbuf  = qbuf  + S;                // [B,H,N,D]
    _Float16* vtbuf = kbuf  + S;                // [B,H,D,N]
    _Float16* aobuf = vtbuf + S;                // [B,N,E]
    _Float16* xh    = aobuf + S;                // [B*N,E] f16 copy of x
    _Float16* wqh   = xh    + S;                // f16 weights
    _Float16* wkh   = wqh   + WS;
    _Float16* wvh   = wkh   + WS;
    _Float16* woh   = wvh   + WS;

    // Pre-convert activations + weights to f16 (bandwidth-trivial).
    f32_to_f16_kernel<<<(int)(S / 8 / 256), 256, 0, stream>>>(x,  xh,  (int)(S / 8));
    f32_to_f16_kernel<<<(int)(WS / 8 / 256), 256, 0, stream>>>(Wq, wqh, (int)(WS / 8));
    f32_to_f16_kernel<<<(int)(WS / 8 / 256), 256, 0, stream>>>(Wk, wkh, (int)(WS / 8));
    f32_to_f16_kernel<<<(int)(WS / 8 / 256), 256, 0, stream>>>(Wv, wvh, (int)(WS / 8));
    f32_to_f16_kernel<<<(int)(WS / 8 / 256), 256, 0, stream>>>(Wo, woh, (int)(WS / 8));

    dim3 gproj(16, 32), bproj(256);
    qkv_proj_kernel<0><<<gproj, bproj, 0, stream>>>(xh, wqh, bq, qbuf);
    qkv_proj_kernel<0><<<gproj, bproj, 0, stream>>>(xh, wkh, bk, kbuf);
    qkv_proj_kernel<1><<<gproj, bproj, 0, stream>>>(xh, wvh, bv, vtbuf);

    attn_kernel<<<dim3(32, 16, 4), dim3(128), 0, stream>>>(qbuf, kbuf, vtbuf, aobuf);

    out_proj_kernel<<<gproj, bproj, 0, stream>>>(aobuf, woh, out);
}
